// TransformerEncoderLayer_78211354460906
// MI455X (gfx1250) — compile-verified, matrix operand order checked by reading, not run on hardware
//
#include <hip/hip_runtime.h>
#include <math.h>

// ---------------------------------------------------------------------------
// Types for CDNA5 WMMA (wave32) + TDM
// ---------------------------------------------------------------------------
typedef __attribute__((ext_vector_type(16))) __bf16        v16bf;
typedef __attribute__((ext_vector_type(8)))  __bf16        v8bf;
typedef __attribute__((ext_vector_type(8)))  float         v8f;
typedef __attribute__((ext_vector_type(4)))  unsigned int  v4u;
typedef __attribute__((ext_vector_type(8)))  int           v8i;
typedef __attribute__((ext_vector_type(4)))  int           v4i;

// Problem constants (match reference setup_inputs)
#define BB   8
#define SS   1024
#define DD   768
#define HH   12
#define DKK  64
#define FF   3072
#define MM   (BB * SS)        // 8192
#define NQKV (HH * 3 * DKK)   // 2304

// float -> bf16, round-to-nearest-even
__device__ __forceinline__ unsigned short f2bf(float f) {
    unsigned int u = __float_as_uint(f);
    u += 0x7FFFu + ((u >> 16) & 1u);
    return (unsigned short)(u >> 16);
}

// Build a 16x32 bf16 WMMA fragment for this lane from two contiguous
// 8-element (16-byte) chunks. Per ISA layout (16-bit A 16x32 / B in [N][K]
// form): lane<16 -> K {0..7, 16..23}; lane>=16 -> K {8..15, 24..31}.
__device__ __forceinline__ v16bf frag16(const unsigned short* p0,
                                        const unsigned short* p1) {
    v8bf lo = *(const v8bf*)p0;
    v8bf hi = *(const v8bf*)p1;
    return __builtin_shufflevector(lo, hi,
        0, 1, 2, 3, 4, 5, 6, 7, 8, 9, 10, 11, 12, 13, 14, 15);
}

// ---------------------------------------------------------------------------
// Tensor Data Mover: DMA a 2-D bf16 tile [nrows][rowlen] (row pitch = stride,
// all in elements) from global memory to LDS at byte offset lds_off.
// D# layout per CDNA5 ISA ch.8: group0 {count=1, lds_addr, global_addr,
// type=2}; group1 {data_size=2B, tensor_dim0/1 = tile dims (no OOB),
// tile_dim0/1, tensor_dim0_stride}. Groups 2/3 zero (2-D tile).
// ---------------------------------------------------------------------------
__device__ __forceinline__ void tdm_load_2d(unsigned lds_off,
                                            const void* gptr,
                                            int rowlen, int nrows,
                                            int stride) {
    unsigned long long ga = (unsigned long long)(size_t)gptr;
    v4u g0;
    g0[0] = 1u;                                        // count=1, user mode
    g0[1] = lds_off;                                   // lds_addr (bytes)
    g0[2] = (unsigned)(ga & 0xFFFFFFFFu);              // global_addr[31:0]
    g0[3] = (unsigned)((ga >> 32) & 0x01FFFFFFu)       // global_addr[56:32]
            | 0x80000000u;                             // type=2 ("image")
    v8i g1;
    g1[0] = (int)(1u << 16);                           // data_size=1 -> 2B
    g1[1] = (int)((unsigned)rowlen << 16);             // tensor_dim0[15:0]
    g1[2] = (int)((unsigned)nrows << 16);              // tensor_dim1[15:0]
    g1[3] = (int)((unsigned)rowlen << 16);             // tile_dim0
    g1[4] = nrows;                                     // tile_dim1 (tile_dim2=0)
    g1[5] = stride;                                    // tensor_dim0_stride[31:0]
    g1[6] = 0;
    g1[7] = 0;
    v4i z4 = {0, 0, 0, 0};
#if defined(__clang_major__) && __clang_major__ >= 23
    v8i z8 = {0, 0, 0, 0, 0, 0, 0, 0};
    __builtin_amdgcn_tensor_load_to_lds(g0, g1, z4, z4, z8, 0);
#else
    __builtin_amdgcn_tensor_load_to_lds(g0, g1, z4, z4, 0);
#endif
}

// ---------------------------------------------------------------------------
// Weight convert + transpose: f32 [batch][K][N] -> bf16 [batch][N][K]
// ---------------------------------------------------------------------------
__global__ void convT_kernel(const float* __restrict__ in,
                             unsigned short* __restrict__ out,
                             int K, int N, int total) {
    int i = blockIdx.x * 256 + threadIdx.x;
    if (i >= total) return;
    int n  = i % N;
    int k  = (i / N) % K;
    int bb = i / (N * K);
    out[(size_t)bb * N * K + (size_t)n * K + k] = f2bf(in[i]);
}

// ---------------------------------------------------------------------------
// LayerNorm over last dim (768), one block per row, output bf16
// ---------------------------------------------------------------------------
__global__ __launch_bounds__(256) void ln_kernel(const float* __restrict__ x,
                                                 const float* __restrict__ g,
                                                 const float* __restrict__ bta,
                                                 unsigned short* __restrict__ out) {
    __shared__ float red[18];
    const int row = blockIdx.x;
    const int t = threadIdx.x, lane = t & 31, wid = t >> 5;
    const float* xr = x + (size_t)row * DD;
    float v[3];
    float s1 = 0.f, s2 = 0.f;
    #pragma unroll
    for (int i = 0; i < 3; ++i) {
        v[i] = xr[t + i * 256];
        s1 += v[i];
        s2 += v[i] * v[i];
    }
    #pragma unroll
    for (int m = 1; m < 32; m <<= 1) {
        s1 += __shfl_xor(s1, m, 32);
        s2 += __shfl_xor(s2, m, 32);
    }
    if (lane == 0) { red[wid] = s1; red[8 + wid] = s2; }
    __syncthreads();
    if (t == 0) {
        float a = 0.f, b = 0.f;
        #pragma unroll
        for (int w = 0; w < 8; ++w) { a += red[w]; b += red[8 + w]; }
        float mu = a * (1.0f / DD);
        float var = b * (1.0f / DD) - mu * mu;
        red[16] = mu;
        red[17] = rsqrtf(var + 1e-5f);
    }
    __syncthreads();
    float mu = red[16], rs = red[17];
    #pragma unroll
    for (int i = 0; i < 3; ++i) {
        int c = t + i * 256;
        out[(size_t)row * DD + c] = f2bf((v[i] - mu) * rs * g[c] + bta[c]);
    }
}

// ---------------------------------------------------------------------------
// Tiled WMMA GEMM with double-buffered TDM tile staging.
// C[M][N] = A[M][K](bf16) * Bt[N][K](bf16)^T  (+epilogue)
// Block tile 128x128, K step 32, 8 waves (2x4), wave tile 64x32.
// Wave 0 drives the Tensor Data Mover; DMA of tile i+1 overlaps WMMA on i.
// ---------------------------------------------------------------------------
template <bool BIAS, bool GELU_, bool RES, bool OUTF, bool OUTB>
__global__ __launch_bounds__(256) void gemm_bf16_kernel(
        const unsigned short* __restrict__ A,
        const unsigned short* __restrict__ Bt,
        const float* __restrict__ bias,
        const float* __restrict__ resid,
        float* __restrict__ outF,
        unsigned short* __restrict__ outB,
        int N, int K) {
    __shared__ unsigned short As[2][128 * 32];
    __shared__ unsigned short Bs[2][128 * 32];

    const int t = threadIdx.x;
    const int lane = t & 31, wid = t >> 5;
    const int wm = wid >> 2, wn = wid & 3;
    const int m0 = blockIdx.y * 128;
    const int n0 = blockIdx.x * 128;
    const int l16 = lane & 15;
    const int koff = (lane >> 4) * 8;

    const unsigned short* Arow = A + (size_t)m0 * K;
    const unsigned short* Brow = Bt + (size_t)n0 * K;

    v8f acc[4][2];
    #pragma unroll
    for (int mi = 0; mi < 4; ++mi)
        #pragma unroll
        for (int ni = 0; ni < 2; ++ni)
            #pragma unroll
            for (int e = 0; e < 8; ++e)
                acc[mi][ni][e] = 0.0f;

    const int niter = K / 32;
    auto issue_tile = [&](int buf, int k0) {
        tdm_load_2d((unsigned)(size_t)(void*)&As[buf][0], Arow + k0, 32, 128, K);
        tdm_load_2d((unsigned)(size_t)(void*)&Bs[buf][0], Brow + k0, 32, 128, K);
    };

    if (wid == 0) issue_tile(0, 0);

    for (int i = 0; i < niter; ++i) {
        if (wid == 0) {
            if (i + 1 < niter) {
                issue_tile((i + 1) & 1, (i + 1) * 32);
                __builtin_amdgcn_s_wait_tensorcnt(2);  // tile i complete
            } else {
                __builtin_amdgcn_s_wait_tensorcnt(0);
            }
        }
        __syncthreads();

        const unsigned short* as = &As[i & 1][0];
        const unsigned short* bs = &Bs[i & 1][0];
        v16bf af[4], bfr[2];
        #pragma unroll
        for (int mi = 0; mi < 4; ++mi) {
            int r = wm * 64 + mi * 16 + l16;
            af[mi] = frag16(&as[r * 32 + koff], &as[r * 32 + koff + 16]);
        }
        #pragma unroll
        for (int ni = 0; ni < 2; ++ni) {
            int c = wn * 32 + ni * 16 + l16;
            bfr[ni] = frag16(&bs[c * 32 + koff], &bs[c * 32 + koff + 16]);
        }
        #pragma unroll
        for (int mi = 0; mi < 4; ++mi)
            #pragma unroll
            for (int ni = 0; ni < 2; ++ni)
                acc[mi][ni] = __builtin_amdgcn_wmma_f32_16x16x32_bf16(
                    false, af[mi], false, bfr[ni], (short)0, acc[mi][ni],
                    false, false);
        __syncthreads();  // all reads done before buffer is overwritten
    }

    const int rbase = (lane >> 4) * 8;
    #pragma unroll
    for (int mi = 0; mi < 4; ++mi)
        #pragma unroll
        for (int ni = 0; ni < 2; ++ni) {
            int col = n0 + wn * 32 + ni * 16 + l16;
            float bcol = BIAS ? bias[col] : 0.0f;
            #pragma unroll
            for (int r = 0; r < 8; ++r) {
                int row = m0 + wm * 64 + mi * 16 + rbase + r;
                float v = acc[mi][ni][r];
                if (BIAS) v += bcol;
                if (GELU_) v = 0.5f * v * (1.0f + erff(v * 0.70710678f));
                size_t idx = (size_t)row * N + col;
                if (RES) v += resid[idx];
                if (OUTF) outF[idx] = v;
                if (OUTB) outB[idx] = f2bf(v);
            }
        }
}

// ---------------------------------------------------------------------------
// Flash attention: one block per (b, h, 128-query tile); 8 waves, each wave
// owns 16 query rows; streams 32-key chunks of K/V through LDS.
// K chunk arrives via TDM; V chunk is transposed in flight by the threads.
// Row sums of P come from an extra WMMA against a ones matrix (replaces a
// 16-lane shuffle reduction); row max keeps the shuffle tree.
// qkv: bf16 [8192][2304] per-head cols [q(64) | k(64) | v(64)].
// z  : bf16 [8192][768]  (heads concatenated)
// ---------------------------------------------------------------------------
__global__ __launch_bounds__(256) void attn_kernel(
        const unsigned short* __restrict__ qkv,
        unsigned short* __restrict__ z) {
    __shared__ unsigned short Kt[32 * 64];      // [key][dk]  == B[N][K] for QK^T
    __shared__ unsigned short Vt[64 * 32];      // [dk][key]  == B[N][K] for P@V
    __shared__ unsigned short Pt[8 * 16 * 32];  // per-wave P staging

    const int t = threadIdx.x, lane = t & 31, wid = t >> 5;
    const int blk = blockIdx.x;
    const int qt = blk & 7;
    const int h  = (blk >> 3) % HH;
    const int b  = blk / (8 * HH);
    const int l16 = lane & 15;
    const int koff = (lane >> 4) * 8;
    const int rbase = (lane >> 4) * 8;
    const int qrow0 = b * SS + qt * 128 + wid * 16;  // global row in [8192]

    // Q fragments (dk=64 -> two 16x32 A-frags), kept in registers
    v16bf qf[2];
    {
        const unsigned short* qp =
            qkv + (size_t)(qrow0 + l16) * NQKV + h * 192;
        qf[0] = frag16(qp + koff, qp + koff + 16);
        qf[1] = frag16(qp + 32 + koff, qp + 32 + koff + 16);
    }

    // ones B-matrix fragment (bf16 1.0 splat) for WMMA row sums
    v16bf ones;
    #pragma unroll
    for (int e = 0; e < 16; ++e)
        ones[e] = __builtin_bit_cast(__bf16, (unsigned short)0x3F80);

    v8f o[4];
    #pragma unroll
    for (int nt = 0; nt < 4; ++nt)
        #pragma unroll
        for (int e = 0; e < 8; ++e) o[nt][e] = 0.0f;

    float mrow[8], lrow[8];
    #pragma unroll
    for (int r = 0; r < 8; ++r) { mrow[r] = -3.402823466e38f; lrow[r] = 0.0f; }

    unsigned short* Pw = &Pt[wid * 16 * 32];

    for (int s0 = 0; s0 < SS; s0 += 32) {
        __syncthreads();
        // K chunk [32 keys][64 dk] via Tensor Data Mover
        if (wid == 0)
            tdm_load_2d((unsigned)(size_t)(void*)Kt,
                        qkv + (size_t)(b * SS + s0) * NQKV + h * 192 + 64,
                        64, 32, NQKV);
        // V chunk transposed to [dk][key] by the threads (TDM cannot transpose)
        {
            int key = t >> 3;
            int kc = (t & 7) * 8;
            uint4 vv = *(const uint4*)
                &qkv[(size_t)(b * SS + s0 + key) * NQKV + h * 192 + 128 + kc];
            unsigned short* vs = (unsigned short*)&vv;
            #pragma unroll
            for (int i = 0; i < 8; ++i) Vt[(kc + i) * 32 + key] = vs[i];
        }
        if (wid == 0) __builtin_amdgcn_s_wait_tensorcnt(0);
        __syncthreads();

        // S = (Q K^T) * scale  -> two 16x16 C-frags over 32 keys
        v8f sf[2];
        #pragma unroll
        for (int ni = 0; ni < 2; ++ni) {
            #pragma unroll
            for (int e = 0; e < 8; ++e) sf[ni][e] = 0.0f;
            #pragma unroll
            for (int ks = 0; ks < 2; ++ks) {
                v16bf kb = frag16(&Kt[(ni * 16 + l16) * 64 + ks * 32 + koff],
                                  &Kt[(ni * 16 + l16) * 64 + ks * 32 + koff + 16]);
                sf[ni] = __builtin_amdgcn_wmma_f32_16x16x32_bf16(
                    false, qf[ks], false, kb, (short)0, sf[ni], false, false);
            }
            #pragma unroll
            for (int e = 0; e < 8; ++e) sf[ni][e] *= 0.125f;  // 1/sqrt(64)
        }

        // row max across the 16 lanes of each half (rows rbase..rbase+7)
        float mx[8];
        #pragma unroll
        for (int r = 0; r < 8; ++r) mx[r] = fmaxf(sf[0][r], sf[1][r]);
        #pragma unroll
        for (int m = 1; m < 16; m <<= 1)
            #pragma unroll
            for (int r = 0; r < 8; ++r)
                mx[r] = fmaxf(mx[r], __shfl_xor(mx[r], m, 32));

        float alpha[8];
        #pragma unroll
        for (int r = 0; r < 8; ++r) {
            float nm = fmaxf(mrow[r], mx[r]);
            alpha[r] = __expf(mrow[r] - nm);
            mrow[r] = nm;
        }

        #pragma unroll
        for (int ni = 0; ni < 2; ++ni)
            #pragma unroll
            for (int r = 0; r < 8; ++r)
                sf[ni][r] = __expf(sf[ni][r] - mrow[r]);

        #pragma unroll
        for (int nt = 0; nt < 4; ++nt)
            #pragma unroll
            for (int r = 0; r < 8; ++r) o[nt][r] *= alpha[r];

        // C-layout -> A-layout conversion for P via wave-private LDS
        #pragma unroll
        for (int ni = 0; ni < 2; ++ni)
            #pragma unroll
            for (int r = 0; r < 8; ++r)
                Pw[(rbase + r) * 32 + ni * 16 + l16] = f2bf(sf[ni][r]);
        // per-wave LDS ops are in-order; compiler inserts dscnt waits
        v16bf pf = frag16(&Pw[l16 * 32 + koff], &Pw[l16 * 32 + koff + 16]);

        // row sums of P via WMMA against ones: each lane's C-frag element r
        // holds sum_k P[rbase+r][k] (identical across the 16 cols)
        v8f srow;
        #pragma unroll
        for (int e = 0; e < 8; ++e) srow[e] = 0.0f;
        srow = __builtin_amdgcn_wmma_f32_16x16x32_bf16(
            false, pf, false, ones, (short)0, srow, false, false);
        #pragma unroll
        for (int r = 0; r < 8; ++r) lrow[r] = lrow[r] * alpha[r] + srow[r];

        // O += P @ V   (K=32 keys in one WMMA step; 4 N-tiles over dk=64)
        #pragma unroll
        for (int nt = 0; nt < 4; ++nt) {
            v16bf vf = frag16(&Vt[(nt * 16 + l16) * 32 + koff],
                              &Vt[(nt * 16 + l16) * 32 + koff + 16]);
            o[nt] = __builtin_amdgcn_wmma_f32_16x16x32_bf16(
                false, pf, false, vf, (short)0, o[nt], false, false);
        }
    }

    // normalize and emit z (bf16, heads concatenated along cols)
    float inv[8];
    #pragma unroll
    for (int r = 0; r < 8; ++r) inv[r] = 1.0f / lrow[r];
    #pragma unroll
    for (int nt = 0; nt < 4; ++nt)
        #pragma unroll
        for (int r = 0; r < 8; ++r) {
            size_t row = (size_t)(qrow0 + rbase + r);
            z[row * DD + h * DKK + nt * 16 + l16] = f2bf(o[nt][r] * inv[r]);
        }
}

// ---------------------------------------------------------------------------
// Host-side orchestration
// ---------------------------------------------------------------------------
extern "C" void kernel_launch(void* const* d_in, const int* in_sizes, int n_in,
                              void* d_out, int out_size, void* d_ws, size_t ws_size,
                              hipStream_t stream) {
    (void)in_sizes; (void)n_in; (void)out_size; (void)ws_size;
    const float* x      = (const float*)d_in[0];
    const float* W_qkv  = (const float*)d_in[1];
    const float* W_o    = (const float*)d_in[2];
    const float* ln1_g  = (const float*)d_in[3];
    const float* ln1_b  = (const float*)d_in[4];
    const float* ln2_g  = (const float*)d_in[5];
    const float* ln2_b  = (const float*)d_in[6];
    const float* fc1_w  = (const float*)d_in[7];
    const float* fc1_b  = (const float*)d_in[8];
    const float* fc2_w  = (const float*)d_in[9];
    const float* fc2_b  = (const float*)d_in[10];
    float* out = (float*)d_out;

    // workspace carve-up (256B aligned regions)
    char* w = (char*)d_ws;
    size_t off = 0;
    auto take = [&](size_t bytes) -> char* {
        char* p = w + off;
        off += (bytes + 255) & ~(size_t)255;
        return p;
    };
    unsigned short* wqkvT = (unsigned short*)take((size_t)NQKV * DD * 2);
    unsigned short* woT   = (unsigned short*)take((size_t)DD * DD * 2);
    unsigned short* fc1T  = (unsigned short*)take((size_t)FF * DD * 2);
    unsigned short* fc2T  = (unsigned short*)take((size_t)DD * FF * 2);
    unsigned short* actA  = (unsigned short*)take((size_t)MM * DD * 2);   // xn -> z -> xn2
    unsigned short* bigB  = (unsigned short*)take((size_t)MM * FF * 2);   // qkv -> h
    float*          x1    = (float*)take((size_t)MM * DD * 4);

    // 1) weight convert+transpose to bf16 [N][K]
    {
        int tot;
        tot = HH * DD * (3 * DKK);
        convT_kernel<<<(tot + 255) / 256, 256, 0, stream>>>(W_qkv, wqkvT, DD, 3 * DKK, tot);
        tot = DD * DD;
        convT_kernel<<<(tot + 255) / 256, 256, 0, stream>>>(W_o, woT, DD, DD, tot);
        tot = DD * FF;
        convT_kernel<<<(tot + 255) / 256, 256, 0, stream>>>(fc1_w, fc1T, DD, FF, tot);
        tot = FF * DD;
        convT_kernel<<<(tot + 255) / 256, 256, 0, stream>>>(fc2_w, fc2T, FF, DD, tot);
    }

    // 2) LN1: x -> xn (bf16)
    ln_kernel<<<MM, 256, 0, stream>>>(x, ln1_g, ln1_b, actA);

    // 3) QKV projection: [8192,768] x [768,2304] -> bf16 qkv
    gemm_bf16_kernel<false, false, false, false, true>
        <<<dim3(NQKV / 128, MM / 128), 256, 0, stream>>>(
            actA, wqkvT, nullptr, nullptr, nullptr, bigB, NQKV, DD);

    // 4) flash attention -> z (bf16, overwrites xn region)
    attn_kernel<<<BB * HH * (SS / 128), 256, 0, stream>>>(bigB, actA);

    // 5) out-proj + residual: x1 = x + z @ W_o
    gemm_bf16_kernel<false, false, true, true, false>
        <<<dim3(DD / 128, MM / 128), 256, 0, stream>>>(
            actA, woT, nullptr, x, x1, nullptr, DD, DD);

    // 6) LN2: x1 -> xn2 (bf16, reuse actA)
    ln_kernel<<<MM, 256, 0, stream>>>(x1, ln2_g, ln2_b, actA);

    // 7) FC1 + bias + exact GELU -> h (bf16, reuse bigB)
    gemm_bf16_kernel<true, true, false, false, true>
        <<<dim3(FF / 128, MM / 128), 256, 0, stream>>>(
            actA, fc1T, fc1_b, nullptr, nullptr, bigB, FF, DD);

    // 8) FC2 + bias + residual -> d_out (f32)
    gemm_bf16_kernel<true, false, true, true, false>
        <<<dim3(DD / 128, MM / 128), 256, 0, stream>>>(
            bigB, fc2T, fc2_b, x1, out, nullptr, DD, FF);
}
